// Net_SDE_Revised_15384572854415
// MI455X (gfx1250) — matrix operand order repaired and checked by reading, not compiled, not run on hardware
//
#include <hip/hip_runtime.h>
#include <hip/hip_bf16.h>

typedef __bf16 v16bf __attribute__((ext_vector_type(16)));
typedef __bf16 v8bf  __attribute__((ext_vector_type(8)));
typedef float  v8f   __attribute__((ext_vector_type(8)));

#define M_TOTAL 8192
#define NSTEPS  20
#define WIDTH   512
#define NNETS   5
#define TILE_M  128          // rows per block
#define BLK     512          // threads per block = 16 waves
#define LDS_STR 520          // 512 + 8 bf16 pad -> 1040B row stride (16B aligned), conflict-free
#define NOPTS   64

struct NetParams {
  const float* W1[NNETS];
  const float* b1[NNETS];
  const float* W2[NNETS];
  const float* b2[NNETS];
  const float* W3[NNETS];
  const float* b3[NNETS];
};

// ---------------------------------------------------------------------------
// One-time (per call) conversion: W2 (512x512 fp32, [k][n]) -> W2T bf16 [n][k]
// column-major so each WMMA B-fragment lane reads 32 contiguous bytes.
// ---------------------------------------------------------------------------
__global__ __launch_bounds__(256) void convert_w2_kernel(NetParams P, __bf16* __restrict__ W2T) {
  int g = blockIdx.y;
  int i = blockIdx.x * blockDim.x + threadIdx.x;     // 0 .. 512*512-1
  int n = i >> 9;
  int k = i & (WIDTH - 1);
  const float* W2 = P.W2[g];
  W2T[(size_t)g * WIDTH * WIDTH + (size_t)n * WIDTH + k] = (__bf16)W2[(size_t)k * WIDTH + n];
}

// ---------------------------------------------------------------------------
// Init S, V, S_hist[0]
// ---------------------------------------------------------------------------
__global__ __launch_bounds__(256) void init_kernel(const float* __restrict__ S0p,
                                                   const float* __restrict__ V0p,
                                                   float* __restrict__ S, float* __restrict__ V,
                                                   float* __restrict__ S_hist) {
  int m = blockIdx.x * blockDim.x + threadIdx.x;
  if (m >= M_TOTAL) return;
  float s0 = S0p[0], v0 = V0p[0];
  S[m] = s0;
  V[m] = v0;
  S_hist[m] = s0;   // row 0
}

// ---------------------------------------------------------------------------
// Fused per-step MLP for net g (blockIdx.y), 128-row M tile (blockIdx.x):
//   h1 = relu([t,S,V,rate] @ W1 + b1)   (VALU, bf16 to LDS)
//   h2 = relu(h1 @ W2 + b2)             (WMMA bf16->f32; 16 acc tiles/wave,
//                                        B double-buffered from L2-resident W2T)
//   out = h2 @ W3 + b3                  (LDS tree reduction)
// ---------------------------------------------------------------------------
__global__ __launch_bounds__(BLK) void step_mlp_kernel(NetParams P,
                                                       const __bf16* __restrict__ W2T,
                                                       const float* __restrict__ S,
                                                       const float* __restrict__ V,
                                                       const float* __restrict__ rate_ptr,
                                                       int step,
                                                       float* __restrict__ out5) {
  extern __shared__ __align__(16) __bf16 tile[];   // TILE_M * LDS_STR bf16 (133 KB)
  __shared__ float sS[TILE_M], sV[TILE_M];
  __shared__ float sred[TILE_M][4];

  const int g    = blockIdx.y;
  const int r0   = blockIdx.x * TILE_M;
  const int tid  = threadIdx.x;
  const int lane = tid & 31;
  const int wave = tid >> 5;           // 0..15, wave owns 32 output columns

  const float* W1  = P.W1[g];
  const float* b1  = P.b1[g];
  const float* b2  = P.b2[g];
  const float* W3  = P.W3[g];
  const float* b3p = P.b3[g];
  const __bf16* W2t = W2T + (size_t)g * WIDTH * WIDTH;

  if (tid < TILE_M) { sS[tid] = S[r0 + tid]; sV[tid] = V[r0 + tid]; }
  __syncthreads();

  const float rate = rate_ptr[0];
  const float t    = (float)step * (1.0f / 360.0f);

  // ---- layer 1 (4 -> 512): one column per thread, 128 rows ----
  {
    const int j = tid;              // BLK == WIDTH
    float w0 = W1[0 * WIDTH + j];
    float w1 = W1[1 * WIDTH + j];
    float w2 = W1[2 * WIDTH + j];
    float w3 = W1[3 * WIDTH + j];
    float base = fmaf(w0, t, fmaf(w3, rate, b1[j]));
    #pragma unroll 4
    for (int m = 0; m < TILE_M; ++m) {
      float v = fmaf(w1, sS[m], fmaf(w2, sV[m], base));
      tile[m * LDS_STR + j] = (__bf16)fmaxf(v, 0.0f);
    }
  }
  __syncthreads();

  // ---- layer 2 (512 -> 512) via WMMA bf16, 8 M-tiles x 2 N-tiles per wave ----
  const int n0    = wave * 32;
  const int arow  = lane & 15;
  const int koffA = (lane < 16) ? 0 : 16;   // A: lanes 0-15 K lo half, 16-31 K hi half (matches B)
  const int bcol  = n0 + (lane & 15);
  const int koffB = (lane < 16) ? 0 : 16;   // B: lanes 0-15 K lo half, 16-31 K hi half

  v8f acc[8][2] = {};

  auto loadB = [&](int k0, v16bf b[2]) {
    #pragma unroll
    for (int nt = 0; nt < 2; ++nt)
      b[nt] = *(const v16bf*)(W2t + (size_t)(bcol + nt * 16) * WIDTH + (k0 + koffB));
  };
  auto doChunk = [&](int k0, const v16bf b[2]) {
    #pragma unroll
    for (int mt = 0; mt < 8; ++mt) {
      const __bf16* ap = &tile[(mt * 16 + arow) * LDS_STR + k0 + koffA];
      v8bf lo = *(const v8bf*)ap;
      v8bf hi = *(const v8bf*)(ap + 8);
      v16bf av;
      #pragma unroll
      for (int e = 0; e < 8; ++e) { av[e] = lo[e]; av[e + 8] = hi[e]; }
      acc[mt][0] = __builtin_amdgcn_wmma_f32_16x16x32_bf16(
          false, av, false, b[0], (short)0, acc[mt][0], false, false);
      acc[mt][1] = __builtin_amdgcn_wmma_f32_16x16x32_bf16(
          false, av, false, b[1], (short)0, acc[mt][1], false, false);
    }
  };

  v16bf bcur[2];
  loadB(0, bcur);
  for (int k0 = 0; k0 < WIDTH - 32; k0 += 32) {
    v16bf bnxt[2];
    loadB(k0 + 32, bnxt);          // prefetch next K-chunk while WMMAs run
    doChunk(k0, bcur);
    bcur[0] = bnxt[0];
    bcur[1] = bnxt[1];
  }
  doChunk(WIDTH - 32, bcur);

  __syncthreads();  // all waves done reading layer-1 tile; reuse it for h2

  // ---- bias + ReLU, write h2 back to LDS (bf16) per C-matrix layout ----
  const int rhalf = (lane < 16) ? 0 : 8;
  #pragma unroll
  for (int mt = 0; mt < 8; ++mt) {
    #pragma unroll
    for (int nt = 0; nt < 2; ++nt) {
      int col = n0 + nt * 16 + (lane & 15);
      float bb = b2[col];
      #pragma unroll
      for (int i = 0; i < 8; ++i) {
        float v = acc[mt][nt][i] + bb;
        tile[(mt * 16 + rhalf + i) * LDS_STR + col] = (__bf16)fmaxf(v, 0.0f);
      }
    }
  }
  __syncthreads();

  // ---- layer 3 (512 -> 1): 4 partial dot-products per row, tree-combine ----
  const int row  = tid >> 2;   // 0..127
  const int part = tid & 3;    // 0..3, 128 columns each
  float s = 0.0f;
  const int jb = part * 128;
  #pragma unroll 8
  for (int j = jb; j < jb + 128; ++j)
    s = fmaf((float)tile[row * LDS_STR + j], W3[j], s);
  sred[row][part] = s;
  __syncthreads();
  if (part == 0) {
    float tot = sred[row][0] + sred[row][1] + sred[row][2] + sred[row][3];
    out5[(size_t)g * M_TOTAL + (r0 + row)] = tot + b3p[0];
  }
}

// ---------------------------------------------------------------------------
// Euler update of (S, V) + record S_hist[step+1]
// ---------------------------------------------------------------------------
__global__ __launch_bounds__(256) void update_kernel(const float* __restrict__ out5,
                                                     const float* __restrict__ z,
                                                     const float* __restrict__ z1,
                                                     float* __restrict__ S, float* __restrict__ V,
                                                     float* __restrict__ S_hist, int step) {
  int m = blockIdx.x * blockDim.x + threadIdx.x;
  if (m >= M_TOTAL) return;
  const float h     = 1.0f / 360.0f;
  const float sqrth = 0.052704627669472994f;  // sqrt(1/360)
  float dW  = sqrth * z [(size_t)m * NSTEPS + step];
  float dW1 = sqrth * z1[(size_t)m * NSTEPS + step];
  float s = S[m], v = V[m];
  float oS    = out5[0 * (size_t)M_TOTAL + m];
  float oD    = out5[1 * (size_t)M_TOTAL + m];
  float oDv   = out5[2 * (size_t)M_TOTAL + m];
  float oDvv  = out5[3 * (size_t)M_TOTAL + m];
  float oDvv1 = out5[4 * (size_t)M_TOTAL + m];
  float Sn = fmaf(oS, h, fmaf(oD, dW, s));
  Sn = fmaxf(Sn, 0.0f);
  float Vn = fmaf(oDv, h, fmaf(oDvv, dW, fmaf(oDvv1, dW1, v)));
  S[m] = Sn;
  V[m] = Vn;
  S_hist[(size_t)(step + 1) * M_TOTAL + m] = Sn;
}

// ---------------------------------------------------------------------------
// Discounted call prices per option: mean over 8192 paths
// ---------------------------------------------------------------------------
__global__ __launch_bounds__(256) void payoff_kernel(const float* __restrict__ x,
                                                     const float* __restrict__ S_hist,
                                                     const float* __restrict__ rate_ptr,
                                                     float* __restrict__ out) {
  __shared__ float red[256];
  int o = blockIdx.x;
  float mat    = x[o * 2 + 0];
  float strike = x[o * 2 + 1];
  int idx = (int)mat;
  const float* Srow = S_hist + (size_t)idx * M_TOTAL;
  float s = 0.0f;
  for (int m = threadIdx.x; m < M_TOTAL; m += 256)
    s += fmaxf(Srow[m] - strike, 0.0f);
  red[threadIdx.x] = s;
  __syncthreads();
  for (int st = 128; st > 0; st >>= 1) {
    if (threadIdx.x < st) red[threadIdx.x] += red[threadIdx.x + st];
    __syncthreads();
  }
  if (threadIdx.x == 0)
    out[o] = red[0] * (1.0f / (float)M_TOTAL) * expf(-rate_ptr[0] * mat);
}

// ---------------------------------------------------------------------------
extern "C" void kernel_launch(void* const* d_in, const int* in_sizes, int n_in,
                              void* d_out, int out_size, void* d_ws, size_t ws_size,
                              hipStream_t stream) {
  (void)in_sizes; (void)n_in; (void)out_size; (void)ws_size;

  const float* x        = (const float*)d_in[0];
  const float* z        = (const float*)d_in[1];
  const float* z1       = (const float*)d_in[2];
  const float* S0p      = (const float*)d_in[3];
  const float* V0p      = (const float*)d_in[4];
  const float* rate_ptr = (const float*)d_in[5];

  NetParams P;
  for (int g = 0; g < NNETS; ++g) {
    int base = 6 + g * 6;
    P.W1[g] = (const float*)d_in[base + 0];
    P.b1[g] = (const float*)d_in[base + 1];
    P.W2[g] = (const float*)d_in[base + 2];
    P.b2[g] = (const float*)d_in[base + 3];
    P.W3[g] = (const float*)d_in[base + 4];
    P.b3[g] = (const float*)d_in[base + 5];
  }

  // workspace layout (256B aligned)
  char* ws = (char*)d_ws;
  __bf16* W2T   = (__bf16*)(ws + 0);                     // 5*512*512*2 = 2,621,440
  float*  S     = (float*)(ws + 2621440);                // 32,768
  float*  V     = (float*)(ws + 2654208);                // 32,768
  float*  Shist = (float*)(ws + 2686976);                // 21*8192*4 = 688,128
  float*  out5  = (float*)(ws + 3375104);                // 5*8192*4  = 163,840

  convert_w2_kernel<<<dim3((WIDTH * WIDTH) / 256, NNETS), 256, 0, stream>>>(P, W2T);
  init_kernel<<<dim3(M_TOTAL / 256), 256, 0, stream>>>(S0p, V0p, S, V, Shist);

  const size_t dynLds = (size_t)TILE_M * LDS_STR * sizeof(__bf16);  // 133,120 B
  for (int step = 0; step < NSTEPS; ++step) {
    step_mlp_kernel<<<dim3(M_TOTAL / TILE_M, NNETS), BLK, dynLds, stream>>>(
        P, W2T, S, V, rate_ptr, step, out5);
    update_kernel<<<dim3(M_TOTAL / 256), 256, 0, stream>>>(
        out5, z, z1, S, V, Shist, step);
  }

  payoff_kernel<<<dim3(NOPTS), 256, 0, stream>>>(x, Shist, rate_ptr, (float*)d_out);
}
